// TransformerBlockWithRoPE_3083786519200
// MI455X (gfx1250) — matrix operand order repaired and checked by reading, not compiled, not run on hardware
//
#include <hip/hip_runtime.h>
#include <cstddef>

// ---------------------------------------------------------------------------
// Types for CDNA5 WMMA
// ---------------------------------------------------------------------------
typedef __attribute__((ext_vector_type(16))) __bf16 v16bf;
typedef __attribute__((ext_vector_type(8)))  float  v8f;

union Frag { v16bf v; unsigned u[8]; };

// ---------------------------------------------------------------------------
// CDNA5 Tensor Data Mover (TDM) support
// ---------------------------------------------------------------------------
#if defined(__has_builtin)
#if __has_builtin(__builtin_amdgcn_tensor_load_to_lds) && __has_builtin(__builtin_amdgcn_s_wait_tensorcnt)
#define HAVE_TDM 1
#endif
#endif

#ifdef HAVE_TDM
typedef __attribute__((ext_vector_type(4))) unsigned u32x4;
typedef __attribute__((ext_vector_type(4))) int      i32x4;
typedef __attribute__((ext_vector_type(8))) int      i32x8;

// 2-D tile load Global -> LDS via TDM (D# per CDNA5 ISA ch.8).
// tileX/tensorX/stride in elements of 2 bytes. padCode: pad 1 dword after
// every 2^(padCode+1) dwords of a row (ISA pad_interval encoding).
__device__ __forceinline__ void tdm_load_2d(unsigned ldsOff, const void* gptr,
                                            unsigned tileX, unsigned tileY,
                                            unsigned tensorX, unsigned rows,
                                            unsigned strideElems, unsigned padCode) {
    unsigned long long ga = (unsigned long long)(size_t)gptr;
    u32x4 g0;
    g0[0] = 1u;                                    // count=1 (valid user D#)
    g0[1] = ldsOff;                                // lds_addr (bytes)
    g0[2] = (unsigned)(ga & 0xffffffffu);          // global_addr[31:0]
    g0[3] = (unsigned)((ga >> 32) & 0x01ffffffu)   // global_addr[56:32]
          | (2u << 30);                            // type = 2 ("image")
    i32x8 g1;
    g1[0] = (int)((1u << 16)                       // data_size = 2 bytes
          | (1u << 20)                             // pad_enable
          | (padCode << 22));                      // pad_interval (amount=1 dw)
    g1[1] = (int)((tensorX & 0xffffu) << 16);      // tensor_dim0[15:0]
    g1[2] = (int)(((tensorX >> 16) & 0xffffu) | ((rows & 0xffffu) << 16));
    g1[3] = (int)(((rows >> 16) & 0xffffu) | ((tileX & 0xffffu) << 16));
    g1[4] = (int)(tileY & 0xffffu);                // tile_dim1 (tile_dim2=0)
    g1[5] = (int)strideElems;                      // tensor_dim0_stride[31:0]
    g1[6] = 0;
    g1[7] = 0;
    i32x4 gz = {0, 0, 0, 0};
#if __clang_major__ >= 23
    i32x8 gz8 = {0, 0, 0, 0, 0, 0, 0, 0};
    __builtin_amdgcn_tensor_load_to_lds(g0, g1, gz, gz, gz8, 0);
#else
    __builtin_amdgcn_tensor_load_to_lds(g0, g1, gz, gz, 0);
#endif
}
#endif  // HAVE_TDM

__device__ __forceinline__ unsigned short f2bf(float f) {
    unsigned u = __builtin_bit_cast(unsigned, f);
    unsigned r = u + 0x7FFFu + ((u >> 16) & 1u);
    return (unsigned short)(r >> 16);
}

__device__ __forceinline__ float gelu_f(float x) {
    float x3 = x * x * x;
    return 0.5f * x * (1.0f + tanhf(0.7978845608028654f * (x + 0.044715f * x3)));
}

__device__ __forceinline__ float rmax16(float x) {
    #pragma unroll
    for (int m = 1; m < 16; m <<= 1) x = fmaxf(x, __shfl_xor(x, m, 32));
    return x;
}
__device__ __forceinline__ float rsum16(float x) {
    #pragma unroll
    for (int m = 1; m < 16; m <<= 1) x += __shfl_xor(x, m, 32);
    return x;
}

// Problem constants (fixed by the reference)
constexpr int Bc = 2, Sc = 2048, Dc = 1024, Hc = 16, Hd = 64, Fc = 4096;
constexpr int Rc = Bc * Sc;          // 4096 token rows

// ---------------------------------------------------------------------------
// Weight conversion: f32 [rows, cols] -> bf16 at dst[row*ld + off + col]
// ---------------------------------------------------------------------------
__global__ void cvt_pack_kernel(const float* __restrict__ src,
                                unsigned short* __restrict__ dst,
                                int cols, int ld, int off, int n) {
    int i = blockIdx.x * blockDim.x + threadIdx.x;
    if (i >= n) return;
    int r = i / cols, c = i % cols;
    dst[(size_t)r * ld + off + c] = f2bf(src[i]);
}

__global__ void pack_bias_kernel(const float* __restrict__ bq,
                                 const float* __restrict__ bk,
                                 const float* __restrict__ bv,
                                 float* __restrict__ dst) {
    int i = blockIdx.x * blockDim.x + threadIdx.x;
    if (i < 1024)      dst[i] = bq[i];
    else if (i < 2048) dst[i] = bk[i - 1024];
    else if (i < 3072) dst[i] = bv[i - 2048];
}

// ---------------------------------------------------------------------------
// LayerNorm: one block (256 thr) per row of D=1024, bf16 out
// ---------------------------------------------------------------------------
__global__ __launch_bounds__(256) void ln_kernel(const float* __restrict__ x,
                                                 const float* __restrict__ g,
                                                 const float* __restrict__ bt,
                                                 unsigned short* __restrict__ y) {
    const int row = blockIdx.x;
    const float* xr = x + (size_t)row * Dc;
    float vals[4], s = 0.f, s2 = 0.f;
    #pragma unroll
    for (int i = 0; i < 4; i++) {
        float v = xr[threadIdx.x + i * 256];
        vals[i] = v; s += v; s2 += v * v;
    }
    #pragma unroll
    for (int m = 1; m < 32; m <<= 1) { s += __shfl_xor(s, m, 32); s2 += __shfl_xor(s2, m, 32); }
    __shared__ float rs[8], rs2[8];
    if ((threadIdx.x & 31) == 0) { rs[threadIdx.x >> 5] = s; rs2[threadIdx.x >> 5] = s2; }
    __syncthreads();
    s = 0.f; s2 = 0.f;
    #pragma unroll
    for (int i = 0; i < 8; i++) { s += rs[i]; s2 += rs2[i]; }
    const float mu  = s * (1.0f / Dc);
    const float var = s2 * (1.0f / Dc) - mu * mu;
    const float inv = rsqrtf(var + 1e-6f);
    unsigned short* yr = y + (size_t)row * Dc;
    #pragma unroll
    for (int i = 0; i < 4; i++) {
        int c = threadIdx.x + i * 256;
        yr[c] = f2bf((vals[i] - mu) * inv * g[c] + bt[c]);
    }
}

// ---------------------------------------------------------------------------
// RoPE + layout change: qkv f32 [R,3072] -> q/k/v bf16 [B,H,S,Hd]
// ---------------------------------------------------------------------------
__global__ __launch_bounds__(256) void rope_kernel(const float* __restrict__ qkv,
                                                   const float* __restrict__ ct,
                                                   const float* __restrict__ st,
                                                   unsigned short* __restrict__ qb,
                                                   unsigned short* __restrict__ kb,
                                                   unsigned short* __restrict__ vb) {
    int t = blockIdx.x * blockDim.x + threadIdx.x;   // over B*S*H*32
    if (t >= Bc * Sc * Hc * 32) return;
    int e = t & 31;
    int h = (t >> 5) & (Hc - 1);
    int s = (t >> 9) & (Sc - 1);
    int b = t >> 20;
    const float c1 = ct[s * Hd + e],       s1 = st[s * Hd + e];
    const float c2 = ct[s * Hd + e + 32],  s2 = st[s * Hd + e + 32];
    size_t base  = ((size_t)(b * Sc + s)) * 3072 + h * Hd;
    size_t obase = (((size_t)(b * Hc + h)) * Sc + s) * Hd;
    float q1 = qkv[base + e],        q2 = qkv[base + e + 32];
    qb[obase + e]      = f2bf(q1 * c1 - q2 * s1);
    qb[obase + e + 32] = f2bf(q2 * c2 + q1 * s2);
    float k1 = qkv[base + 1024 + e], k2 = qkv[base + 1024 + e + 32];
    kb[obase + e]      = f2bf(k1 * c1 - k2 * s1);
    kb[obase + e + 32] = f2bf(k2 * c2 + k1 * s2);
    vb[obase + e]      = f2bf(qkv[base + 2048 + e]);
    vb[obase + e + 32] = f2bf(qkv[base + 2048 + e + 32]);
}

// ---------------------------------------------------------------------------
// Tiled bf16 WMMA GEMM: C[M,N] = A[M,K] * B[K,N] (+bias)(gelu)(+resid)
// Block 256 thr = 8 waves; tile 128x128x32; wave tile 64x32 (4x2 WMMA frags).
// Double-buffered LDS; A-tile moved by TDM (tensor_load_to_lds, pad-inserted
// pitch), B-tile register-staged and pair-packed into WMMA-B layout.
// ---------------------------------------------------------------------------
__global__ __launch_bounds__(256, 1) void gemm_bf16_kernel(
    const unsigned short* __restrict__ A, const unsigned short* __restrict__ B,
    int M, int N, int K,
    const float* __restrict__ bias, const float* __restrict__ resid,
    float* __restrict__ outF, unsigned short* __restrict__ outB, int gelu) {
    __shared__ unsigned lA[2][128][17];     // [m][k-pair], pitch 17 (bank-safe)
    __shared__ unsigned lB[2][16][132];     // [k-pair][n], pitch 132

    const int tid  = threadIdx.x;
    const int lane = tid & 31;
    const int w    = tid >> 5;
    const int wm   = w >> 2;                // 0..1
    const int wn   = w & 3;                 // 0..3
    const int m0   = blockIdx.y * 128;
    const int n0   = blockIdx.x * 128;
    const int lm   = lane & 15;
    const int lh   = lane >> 4;

    const unsigned* Bu = (const unsigned*)B;
    const int Nu = N >> 1;
    const int nT = K >> 5;                  // # of 32-wide k tiles

    const v8f zero8 = {0.f, 0.f, 0.f, 0.f, 0.f, 0.f, 0.f, 0.f};
    v8f acc[4][2];
    #pragma unroll
    for (int i = 0; i < 4; i++)
        #pragma unroll
        for (int j = 0; j < 2; j++) acc[i][j] = zero8;

    const int bkp = tid >> 4, bseg = tid & 15;     // B loader: k-pair + 8-col seg
#ifndef HAVE_TDM
    const unsigned* Au = (const unsigned*)A;
    const int Ku = K >> 1;
    const int arow = tid >> 1, ahalf = tid & 1;    // A loader: row + 8-dword half
#endif

    // ---- prologue: stage tile 0 into buffer 0 ----
#ifdef HAVE_TDM
    if (w == 0)
        tdm_load_2d((unsigned)(size_t)&lA[0][0][0], A + (size_t)m0 * K,
                    /*tileX=*/32, /*tileY=*/128, /*tensorX=*/K, /*rows=*/M,
                    /*stride=*/K, /*padCode=*/3);  // pad 1 dw per 16 dw -> pitch 17
#else
    {
        const unsigned* src = Au + (size_t)(m0 + arow) * Ku + ahalf * 8;
        uint4 p0 = *(const uint4*)src;
        uint4 p1 = *(const uint4*)(src + 4);
        unsigned* d = &lA[0][arow][ahalf * 8];
        d[0] = p0.x; d[1] = p0.y; d[2] = p0.z; d[3] = p0.w;
        d[4] = p1.x; d[5] = p1.y; d[6] = p1.z; d[7] = p1.w;
    }
#endif
    {
        const unsigned* srcE = Bu + (size_t)(2 * bkp) * Nu + bseg * 4;
        uint4 e = *(const uint4*)srcE;
        uint4 o = *(const uint4*)(srcE + Nu);
        unsigned eu[4] = {e.x, e.y, e.z, e.w};
        unsigned ou[4] = {o.x, o.y, o.z, o.w};
        unsigned* d = &lB[0][bkp][bseg * 8];
        #pragma unroll
        for (int j = 0; j < 4; j++) {
            d[2 * j]     = (eu[j] & 0xffffu) | (ou[j] << 16);
            d[2 * j + 1] = (eu[j] >> 16)     | (ou[j] & 0xffff0000u);
        }
    }
#ifdef HAVE_TDM
    if (w == 0) __builtin_amdgcn_s_wait_tensorcnt(0);
#endif
    __syncthreads();

    for (int t = 0; t < nT; ++t) {
        const int buf = t & 1;
        const bool more = (t + 1 < nT);

        // ---- issue prefetch of tile t+1 into buffer buf^1 ----
        uint4 nbe, nbo;
#ifndef HAVE_TDM
        uint4 na0, na1;
#endif
        if (more) {
            const unsigned* srcE =
                Bu + (size_t)((t + 1) * 32 + 2 * bkp) * Nu + bseg * 4;
            nbe = *(const uint4*)srcE;
            nbo = *(const uint4*)(srcE + Nu);
#ifdef HAVE_TDM
            if (w == 0)
                tdm_load_2d((unsigned)(size_t)&lA[buf ^ 1][0][0],
                            A + (size_t)m0 * K + (t + 1) * 32,
                            32, 128, K, M, K, 3);
#else
            const unsigned* srcA =
                Au + (size_t)(m0 + arow) * Ku + ((t + 1) * 16) + ahalf * 8;
            na0 = *(const uint4*)srcA;
            na1 = *(const uint4*)(srcA + 4);
#endif
        }

        // ---- compute on buffer buf ----
        Frag af[4], bf[2];
        #pragma unroll
        for (int mf = 0; mf < 4; mf++) {
            const unsigned* r = &lA[buf][wm * 64 + mf * 16 + lm][lh * 4];
            af[mf].u[0] = r[0];  af[mf].u[1] = r[1];  af[mf].u[2] = r[2];  af[mf].u[3] = r[3];
            af[mf].u[4] = r[8];  af[mf].u[5] = r[9];  af[mf].u[6] = r[10]; af[mf].u[7] = r[11];
        }
        #pragma unroll
        for (int nf = 0; nf < 2; nf++) {
            int n = wn * 32 + nf * 16 + lm;
            #pragma unroll
            for (int vv = 0; vv < 8; vv++) bf[nf].u[vv] = lB[buf][(vv >> 2) * 8 + lh * 4 + (vv & 3)][n];
        }
        #pragma unroll
        for (int mf = 0; mf < 4; mf++)
            #pragma unroll
            for (int nf = 0; nf < 2; nf++)
                acc[mf][nf] = __builtin_amdgcn_wmma_f32_16x16x32_bf16(
                    false, af[mf].v, false, bf[nf].v, (short)0, acc[mf][nf], false, false);

        // ---- store staged tile t+1 ----
        if (more) {
            unsigned eu[4] = {nbe.x, nbe.y, nbe.z, nbe.w};
            unsigned ou[4] = {nbo.x, nbo.y, nbo.z, nbo.w};
            unsigned* d = &lB[buf ^ 1][bkp][bseg * 8];
            #pragma unroll
            for (int j = 0; j < 4; j++) {
                d[2 * j]     = (eu[j] & 0xffffu) | (ou[j] << 16);
                d[2 * j + 1] = (eu[j] >> 16)     | (ou[j] & 0xffff0000u);
            }
#ifndef HAVE_TDM
            unsigned* da = &lA[buf ^ 1][arow][ahalf * 8];
            da[0] = na0.x; da[1] = na0.y; da[2] = na0.z; da[3] = na0.w;
            da[4] = na1.x; da[5] = na1.y; da[6] = na1.z; da[7] = na1.w;
#endif
        }
#ifdef HAVE_TDM
        if (w == 0) __builtin_amdgcn_s_wait_tensorcnt(0);
#endif
        __syncthreads();
    }

    // Epilogue: bias -> gelu -> residual -> store
    #pragma unroll
    for (int mf = 0; mf < 4; mf++) {
        #pragma unroll
        for (int nf = 0; nf < 2; nf++) {
            int col = n0 + wn * 32 + nf * 16 + lm;
            float bv = bias ? bias[col] : 0.0f;
            #pragma unroll
            for (int vv = 0; vv < 8; vv++) {
                int row = m0 + wm * 64 + mf * 16 + lh * 8 + vv;
                float val = acc[mf][nf][vv] + bv;
                if (gelu) val = gelu_f(val);
                size_t idx = (size_t)row * N + col;
                if (resid) val += resid[idx];
                if (outF)  outF[idx] = val;
                if (outB)  outB[idx] = f2bf(val);
            }
        }
    }
}

// ---------------------------------------------------------------------------
// Flash attention: grid (S/128, B*H); 8 waves, each owns a 16-row q fragment.
// q/k/v bf16 [B,H,S,Hd]; out bf16 [B,S,H*Hd]. K tile moved by TDM.
// ---------------------------------------------------------------------------
__global__ __launch_bounds__(256, 1) void attn_kernel(
    const unsigned short* __restrict__ q, const unsigned short* __restrict__ k,
    const unsigned short* __restrict__ v, unsigned short* __restrict__ o) {
    __shared__ unsigned lK[64][33];          // [key][e-pair], pitch 33
    __shared__ unsigned lV[32][66];          // [key-pair][hd], pitch 66
    __shared__ unsigned lP[8][16][33];       // per-wave P transpose staging

    const int tid  = threadIdx.x;
    const int lane = tid & 31;
    const int w    = tid >> 5;
    const int bh   = blockIdx.y;             // b*H + h
    const int q0   = blockIdx.x * 128;
    const int lm   = lane & 15;
    const int lh   = lane >> 4;

    const unsigned short* kp16 = k + (size_t)bh * Sc * Hd;
    const unsigned* qu = (const unsigned*)q + (size_t)bh * Sc * 32;
    const unsigned* vu = (const unsigned*)v + (size_t)bh * Sc * 32;
#ifndef HAVE_TDM
    const unsigned* ku = (const unsigned*)kp16;
#endif

    // Q fragments in WMMA-A layout, held in registers for the whole kernel
    Frag aq[2];
    {
        const unsigned* r = qu + (size_t)(q0 + w * 16 + lm) * 32;
        #pragma unroll
        for (int f = 0; f < 2; f++) {
            aq[f].u[0] = r[f * 16 + lh * 4 + 0]; aq[f].u[1] = r[f * 16 + lh * 4 + 1];
            aq[f].u[2] = r[f * 16 + lh * 4 + 2]; aq[f].u[3] = r[f * 16 + lh * 4 + 3];
            aq[f].u[4] = r[f * 16 + 8 + lh * 4 + 0]; aq[f].u[5] = r[f * 16 + 8 + lh * 4 + 1];
            aq[f].u[6] = r[f * 16 + 8 + lh * 4 + 2]; aq[f].u[7] = r[f * 16 + 8 + lh * 4 + 3];
        }
    }

    const v8f zero8 = {0.f, 0.f, 0.f, 0.f, 0.f, 0.f, 0.f, 0.f};
    v8f Oacc[4];
    float Mx[8], Lx[8];
    #pragma unroll
    for (int j = 0; j < 4; j++) Oacc[j] = zero8;
    #pragma unroll
    for (int i = 0; i < 8; i++) { Mx[i] = -1e30f; Lx[i] = 0.f; }

    const float scale = 0.125f;              // 1/sqrt(64)

    for (int s0 = 0; s0 < Sc; s0 += 64) {
        __syncthreads();
#ifdef HAVE_TDM
        if (w == 0)
            tdm_load_2d((unsigned)(size_t)&lK[0][0], kp16 + (size_t)s0 * Hd,
                        /*tileX=*/64, /*tileY=*/64, /*tensorX=*/Hd, /*rows=*/Sc,
                        /*stride=*/Hd, /*padCode=*/4);  // pad 1 dw per 32 dw
#else
        {   // K tile: 64 keys x 64 e (32 dwords per key)
            int row = tid >> 2, seg = tid & 3;
            const unsigned* src = ku + (size_t)(s0 + row) * 32 + seg * 8;
            uint4 a0 = *(const uint4*)src;
            uint4 a1 = *(const uint4*)(src + 4);
            unsigned* d = &lK[row][seg * 8];
            d[0] = a0.x; d[1] = a0.y; d[2] = a0.z; d[3] = a0.w;
            d[4] = a1.x; d[5] = a1.y; d[6] = a1.z; d[7] = a1.w;
        }
#endif
        {   // V tile packed into WMMA-B pair layout: lV[kp][n]={V[2kp][n],V[2kp+1][n]}
            int kpi = tid >> 3, seg = tid & 7;
            const unsigned* srcE = vu + (size_t)(s0 + 2 * kpi) * 32 + seg * 4;
            uint4 e  = *(const uint4*)srcE;
            uint4 oo = *(const uint4*)(srcE + 32);
            unsigned eu[4] = {e.x, e.y, e.z, e.w};
            unsigned ou[4] = {oo.x, oo.y, oo.z, oo.w};
            unsigned* d = &lV[kpi][seg * 8];
            #pragma unroll
            for (int j = 0; j < 4; j++) {
                d[2 * j]     = (eu[j] & 0xffffu) | (ou[j] << 16);
                d[2 * j + 1] = (eu[j] >> 16)     | (ou[j] & 0xffff0000u);
            }
        }
#ifdef HAVE_TDM
        if (w == 0) __builtin_amdgcn_s_wait_tensorcnt(0);
#endif
        __syncthreads();

        // scores: S = Q (16x64) * K^T (64x64) in 4 key sub-tiles
        v8f sc[4];
        #pragma unroll
        for (int j = 0; j < 4; j++) {
            sc[j] = zero8;
            #pragma unroll
            for (int f = 0; f < 2; f++) {
                Frag bk_;
                int key = j * 16 + lm;
                #pragma unroll
                for (int vv = 0; vv < 8; vv++) bk_.u[vv] = lK[key][f * 16 + (vv >> 2) * 8 + lh * 4 + (vv & 3)];
                sc[j] = __builtin_amdgcn_wmma_f32_16x16x32_bf16(
                    false, aq[f].v, false, bk_.v, (short)0, sc[j], false, false);
            }
        }

        // Online softmax per row; write probs (bf16) into per-wave LDS transpose buf
        unsigned short* pp = (unsigned short*)&lP[w][0][0];
        #pragma unroll
        for (int vv = 0; vv < 8; vv++) {
            float tmax = -1e30f;
            #pragma unroll
            for (int j = 0; j < 4; j++) tmax = fmaxf(tmax, sc[j][vv]);
            tmax = rmax16(tmax) * scale;
            float newM  = fmaxf(Mx[vv], tmax);
            float alpha = __expf(Mx[vv] - newM);
            float ls = 0.f;
            int rowoff = (lh * 8 + vv) * 33 * 2;
            #pragma unroll
            for (int j = 0; j < 4; j++) {
                float p = __expf(sc[j][vv] * scale - newM);
                ls += p;
                int col = j * 16 + lm;
                pp[rowoff + (col >> 1) * 2 + (col & 1)] = f2bf(p);
            }
            ls = rsum16(ls);
            Lx[vv] = Lx[vv] * alpha + ls;
            Mx[vv] = newM;
            #pragma unroll
            for (int j = 0; j < 4; j++) Oacc[j][vv] *= alpha;
        }

        // Read P back in WMMA-A layout (wave-local LDS, DScnt-ordered)
        Frag ap[2];
        {
            const unsigned* r = &lP[w][lm][0];
            #pragma unroll
            for (int f = 0; f < 2; f++) {
                ap[f].u[0] = r[f * 16 + lh * 4 + 0]; ap[f].u[1] = r[f * 16 + lh * 4 + 1];
                ap[f].u[2] = r[f * 16 + lh * 4 + 2]; ap[f].u[3] = r[f * 16 + lh * 4 + 3];
                ap[f].u[4] = r[f * 16 + 8 + lh * 4 + 0]; ap[f].u[5] = r[f * 16 + 8 + lh * 4 + 1];
                ap[f].u[6] = r[f * 16 + 8 + lh * 4 + 2]; ap[f].u[7] = r[f * 16 + 8 + lh * 4 + 3];
            }
        }

        // O += P (16x64) * V (64x64)
        #pragma unroll
        for (int jo = 0; jo < 4; jo++) {
            #pragma unroll
            for (int f = 0; f < 2; f++) {
                Frag bv_;
                int nhd = jo * 16 + lm;
                #pragma unroll
                for (int vv = 0; vv < 8; vv++) bv_.u[vv] = lV[f * 16 + (vv >> 2) * 8 + lh * 4 + (vv & 3)][nhd];
                Oacc[jo] = __builtin_amdgcn_wmma_f32_16x16x32_bf16(
                    false, ap[f].v, false, bv_.v, (short)0, Oacc[jo], false, false);
            }
        }
    }

    // Final normalize + store to [B, S, H*Hd] bf16
    const int b = bh >> 4, h = bh & 15;
    #pragma unroll
    for (int jo = 0; jo < 4; jo++) {
        #pragma unroll
        for (int vv = 0; vv < 8; vv++) {
            int srow = q0 + w * 16 + lh * 8 + vv;
            int col  = jo * 16 + lm;
            float val = Oacc[jo][vv] / Lx[vv];
            o[(((size_t)(b * Sc + srow)) * Hc + h) * Hd + col] = f2bf(val);
        }
    }
}

// ---------------------------------------------------------------------------
// Host orchestration
// ---------------------------------------------------------------------------
extern "C" void kernel_launch(void* const* d_in, const int* in_sizes, int n_in,
                              void* d_out, int out_size, void* d_ws, size_t ws_size,
                              hipStream_t stream) {
    (void)in_sizes; (void)n_in; (void)out_size; (void)ws_size;
    const float* x    = (const float*)d_in[0];
    const float* rc   = (const float*)d_in[1];
    const float* rsn  = (const float*)d_in[2];
    const float* ln1s = (const float*)d_in[3];
    const float* ln1b = (const float*)d_in[4];
    const float* wq   = (const float*)d_in[5];
    const float* bq   = (const float*)d_in[6];
    const float* wk   = (const float*)d_in[7];
    const float* bk   = (const float*)d_in[8];
    const float* wv   = (const float*)d_in[9];
    const float* bv   = (const float*)d_in[10];
    const float* wo   = (const float*)d_in[11];
    const float* bo   = (const float*)d_in[12];
    const float* ln2s = (const float*)d_in[13];
    const float* ln2b = (const float*)d_in[14];
    const float* w1   = (const float*)d_in[15];
    const float* b1   = (const float*)d_in[16];
    const float* w2   = (const float*)d_in[17];
    const float* b2   = (const float*)d_in[18];

    char* ws = (char*)d_ws;
    size_t off = 0;
    auto alloc = [&](size_t bytes) {
        void* p = ws + off;
        off += (bytes + 255) & ~(size_t)255;
        return p;
    };
    float*          qkv    = (float*)alloc((size_t)Rc * 3072 * 4);      // 48 MB
    float*          x2     = (float*)alloc((size_t)Rc * Dc * 4);        // 16 MB
    unsigned short* wqkv_b = (unsigned short*)alloc((size_t)Dc * 3072 * 2);
    unsigned short* wo_b   = (unsigned short*)alloc((size_t)Dc * Dc * 2);
    unsigned short* w1_b   = (unsigned short*)alloc((size_t)Dc * Fc * 2);
    unsigned short* w2_b   = (unsigned short*)alloc((size_t)Fc * Dc * 2);
    float*          bqkv   = (float*)alloc(3072 * 4);
    unsigned short* actb   = (unsigned short*)alloc((size_t)Rc * Dc * 2); // y1 / attn_out / y2
    unsigned short* qb     = (unsigned short*)alloc((size_t)Bc * Hc * Sc * Hd * 2);
    unsigned short* kb     = (unsigned short*)alloc((size_t)Bc * Hc * Sc * Hd * 2);
    unsigned short* vb     = (unsigned short*)alloc((size_t)Bc * Hc * Sc * Hd * 2);
    unsigned short* hb     = (unsigned short*)alloc((size_t)Rc * Fc * 2); // 32 MB

    // --- weight conversion / packing ---
    {
        int n = Dc * Dc;
        int blks = (n + 255) / 256;
        cvt_pack_kernel<<<blks, 256, 0, stream>>>(wq, wqkv_b, 1024, 3072, 0,    n);
        cvt_pack_kernel<<<blks, 256, 0, stream>>>(wk, wqkv_b, 1024, 3072, 1024, n);
        cvt_pack_kernel<<<blks, 256, 0, stream>>>(wv, wqkv_b, 1024, 3072, 2048, n);
        cvt_pack_kernel<<<blks, 256, 0, stream>>>(wo, wo_b,   1024, 1024, 0,    n);
        int n1 = Dc * Fc;
        cvt_pack_kernel<<<(n1 + 255) / 256, 256, 0, stream>>>(w1, w1_b, 4096, 4096, 0, n1);
        cvt_pack_kernel<<<(n1 + 255) / 256, 256, 0, stream>>>(w2, w2_b, 1024, 1024, 0, n1);
        pack_bias_kernel<<<12, 256, 0, stream>>>(bq, bk, bv, bqkv);
    }

    // --- LN1 ---
    ln_kernel<<<Rc, 256, 0, stream>>>(x, ln1s, ln1b, actb);

    // --- QKV projection (fused bias) ---
    gemm_bf16_kernel<<<dim3(3072 / 128, Rc / 128), 256, 0, stream>>>(
        actb, wqkv_b, Rc, 3072, Dc, bqkv, nullptr, qkv, nullptr, 0);

    // --- RoPE + bf16 head-major q/k/v ---
    {
        int n = Bc * Sc * Hc * 32;
        rope_kernel<<<(n + 255) / 256, 256, 0, stream>>>(qkv, rc, rsn, qb, kb, vb);
    }

    // --- Flash attention ---
    attn_kernel<<<dim3(Sc / 128, Bc * Hc), 256, 0, stream>>>(qb, kb, vb, actb);

    // --- Output projection (+bias, +residual x) -> x2 ---
    gemm_bf16_kernel<<<dim3(Dc / 128, Rc / 128), 256, 0, stream>>>(
        actb, wo_b, Rc, Dc, Dc, bo, x, x2, nullptr, 0);

    // --- LN2 ---
    ln_kernel<<<Rc, 256, 0, stream>>>(x2, ln2s, ln2b, actb);

    // --- MLP up (+bias, GELU) -> bf16 hb ---
    gemm_bf16_kernel<<<dim3(Fc / 128, Rc / 128), 256, 0, stream>>>(
        actb, w1_b, Rc, Fc, Dc, b1, nullptr, nullptr, hb, 1);

    // --- MLP down (+bias, +residual x2) -> d_out f32 ---
    gemm_bf16_kernel<<<dim3(Dc / 128, Rc / 128), 256, 0, stream>>>(
        hb, w2_b, Rc, Dc, Fc, b2, x2, (float*)d_out, nullptr, 0);
}
